// SelfAttentionkv_19241453486133
// MI455X (gfx1250) — matrix operand order repaired and checked by reading, not compiled, not run on hardware
//
#include <hip/hip_runtime.h>
#include <hip/hip_bf16.h>
#include <math.h>

// ---------------------------------------------------------------------------
// Types for CDNA5 WMMA (wave32): D(16x16 f32) = A(16x32 bf16) * B(32x16 bf16) + C
// ---------------------------------------------------------------------------
typedef __bf16 bf16_t;
typedef __attribute__((ext_vector_type(16))) __bf16 v16bf;
typedef __attribute__((ext_vector_type(8)))  __bf16 v8bf;
typedef __attribute__((ext_vector_type(8)))  float  v8f;
typedef __attribute__((ext_vector_type(4)))  float  v4f;

union V8F  { v8f v; float f[8]; };
union V16B { v16bf v; __bf16 e[16]; };

__device__ __forceinline__ v8f wmma_bf16(v16bf a, v16bf b, v8f c) {
    return __builtin_amdgcn_wmma_f32_16x16x32_bf16(false, a, false, b, (short)0, c, false, false);
}

// A fragment (16x32 bf16): lane half 0 -> K = k0+[0..15]
//                          lane half 1 -> K = k0+16+[0..15]
__device__ __forceinline__ v16bf load_a_row(const bf16_t* arow, int k0, int half) {
    const bf16_t* p = arow + k0 + half * 16;
    v8bf lo = *(const v8bf*)p;
    v8bf hi = *(const v8bf*)(p + 8);
    return __builtin_shufflevector(lo, hi, 0,1,2,3,4,5,6,7,8,9,10,11,12,13,14,15);
}

__device__ __forceinline__ v16bf load_a_tile(const bf16_t* A, int lda, int m0, int lane, int k0) {
    return load_a_row(A + (size_t)(m0 + (lane & 15)) * lda, k0, (lane >> 4) & 1);
}

// B fragment (32x16 bf16) for B[k][n] = W[n][k], W row-major (ldw = K stride).
__device__ __forceinline__ v16bf load_b_tile(const bf16_t* W, int ldw, int n0, int k0, int lane) {
    int half = (lane >> 4) & 1;
    const bf16_t* p = W + (size_t)(n0 + (lane & 15)) * ldw + k0 + half * 16;
    return *(const v16bf*)p;
}

// ---------------------------------------------------------------------------
// CDNA5 async global->LDS copy (16 bytes per lane), tracked by ASYNCcnt.
// vdst = LDS byte address (low 32 bits of generic pointer), vaddr = global.
// ---------------------------------------------------------------------------
__device__ __forceinline__ void async_b128(const bf16_t* g, bf16_t* l) {
    unsigned int lo = (unsigned int)(unsigned long long)l; // LDS byte offset
    asm volatile("global_load_async_to_lds_b128 %0, %1, off"
                 :: "v"(lo), "v"(g) : "memory");
}
template <int N>
__device__ __forceinline__ void wait_async() {
    asm volatile("s_wait_asynccnt %0" :: "n"(N) : "memory");
}

// ---------------------------------------------------------------------------
// Problem constants
// ---------------------------------------------------------------------------
#define BATCH 4
#define NQ    4608
#define CH    512
#define NH    8
#define DH    64
#define NK    1152          // 48*24 after 2x2/2 conv on 96x48
#define MQ    (BATCH * NQ)  // 18432
#define MK    (BATCH * NK)  // 4608
#define LDP   1156          // attention LDS row stride (floats)

#define BN    256           // block N tile
#define KC    64            // K chunk staged per stage
#define LDB   72            // LDS row stride (bf16) for staged tiles; 72*2=144B
#define GEMM_LDS_BYTES ((2 * BN * LDB + 2 * 16 * LDB) * 2)   // 78336

// ---------------------------------------------------------------------------
// LDS-staged, async-pipelined GEMM core.
// Block: 128 threads (4 waves). Tile: 16(M) x 256(N). K = NTAPS * 512.
// asrc[t]: per-copying-thread global base of A row (row = tid>>3) for tap t.
// wbase[t]: weight base (row-major [n][k]) for tap t.
// ---------------------------------------------------------------------------
template <int NTAPS>
__device__ __forceinline__ void gemm_staged(const bf16_t* const* asrc,
                                            const bf16_t* const* wbase,
                                            int n_blk, int tid,
                                            bf16_t* Ab, bf16_t* Bb, v8f acc[4]) {
    const int lane = tid & 31, wave = tid >> 5, half = (lane >> 4) & 1;
    const int nsub = wave * 64;
    const int crow = tid >> 3, cseg = tid & 7;   // copy-thread row/segment
    constexpr int CPT = CH / KC;                 // chunks per tap (8)
    constexpr int TOTAL = NTAPS * CPT;

    auto issue = [&](int chunk, int s) {
        int tap = chunk / CPT;
        int kb  = (chunk % CPT) * KC;
        // B tile: 256 rows x 64 cols; 16 b128 transfers per thread
        const bf16_t* gsrc = wbase[tap] + (size_t)(n_blk + crow) * CH + kb + cseg * 8;
        bf16_t* ldst = Bb + s * (BN * LDB) + crow * LDB + cseg * 8;
        #pragma unroll
        for (int i = 0; i < 16; i++) {
            async_b128(gsrc, ldst);
            gsrc += 16 * CH;
            ldst += 16 * LDB;
        }
        // A tile: 16 rows x 64 cols; 1 b128 transfer per thread
        async_b128(asrc[tap] + kb + cseg * 8,
                   Ab + s * (16 * LDB) + crow * LDB + cseg * 8);
    };

    issue(0, 0);
    for (int c = 0; c < TOTAL; c++) {
        int s = c & 1;
        if (c + 1 < TOTAL) { issue(c + 1, s ^ 1); wait_async<17>(); }
        else                 wait_async<0>();
        __syncthreads();
        const bf16_t* Abs = Ab + s * (16 * LDB) + (lane & 15) * LDB;
        const bf16_t* Bbs = Bb + s * (BN * LDB);
        #pragma unroll
        for (int ks = 0; ks < KC; ks += 32) {
            v16bf a = load_a_row(Abs, ks, half);
            #pragma unroll
            for (int t = 0; t < 4; t++) {
                const bf16_t* p = Bbs + (size_t)(nsub + t * 16 + (lane & 15)) * LDB
                                  + ks + half * 16;
                acc[t] = wmma_bf16(a, *(const v16bf*)p, acc[t]);
            }
        }
        __syncthreads();
    }
}

// ---------------------------------------------------------------------------
// Conversions / packing
// ---------------------------------------------------------------------------
__global__ void cvt_f32_bf16(const float* __restrict__ in, bf16_t* __restrict__ out, int n) {
    int i = blockIdx.x * blockDim.x + threadIdx.x;
    if (i < n) out[i] = (bf16_t)in[i];
}

// sr_w (C,C,2,2) -> wpack[t][co][ci], t = kh*2+kw
__global__ void pack_srw(const float* __restrict__ w, bf16_t* __restrict__ out, int n) {
    int i = blockIdx.x * blockDim.x + threadIdx.x;
    if (i >= n) return;
    int t  = i >> 18;
    int r  = i & 262143;
    int co = r >> 9;
    int ci = r & 511;
    out[i] = (bf16_t)w[co * 2048 + ci * 4 + t];
}

// ---------------------------------------------------------------------------
// GEMM: q = x @ q_w.T + q_b ; scatter to q_bf[b][h][n][d]
// grid (MQ/16, 2), block 128, dynamic LDS
// ---------------------------------------------------------------------------
__global__ __launch_bounds__(128) void gemm_q(const bf16_t* __restrict__ A,
                                              const bf16_t* __restrict__ W,
                                              const float* __restrict__ bias,
                                              bf16_t* __restrict__ qout) {
    extern __shared__ bf16_t smem_bf[];
    bf16_t* Bb = smem_bf;
    bf16_t* Ab = smem_bf + 2 * BN * LDB;
    int tid = threadIdx.x, lane = tid & 31, wave = tid >> 5;
    int m0 = blockIdx.x * 16;
    int n_blk = blockIdx.y * BN;
    int half = (lane >> 4) & 1;

    const bf16_t* asrc[1] = { A + (size_t)(m0 + (tid >> 3)) * CH };
    const bf16_t* wb[1]   = { W };
    v8f acc[4] = {{0,0,0,0,0,0,0,0},{0,0,0,0,0,0,0,0},{0,0,0,0,0,0,0,0},{0,0,0,0,0,0,0,0}};
    gemm_staged<1>(asrc, wb, n_blk, tid, Ab, Bb, acc);

    int n0 = n_blk + wave * 64;
    int b = m0 / NQ;
    int nbase = m0 % NQ + half * 8;
    int nc = lane & 15;
    #pragma unroll
    for (int t = 0; t < 4; t++) {
        int c = n0 + t * 16 + nc;
        int h = c >> 6, d = c & 63;
        float bv = bias[c];
        V8F u; u.v = acc[t];
        #pragma unroll
        for (int j = 0; j < 8; j++) {
            int n = nbase + j;
            qout[(((size_t)b * NH + h) * NQ + n) * DH + d] = (bf16_t)(u.f[j] + bv);
        }
    }
}

// ---------------------------------------------------------------------------
// SR conv as implicit GEMM over 4 taps x 512 channels
// grid (MK/16, 2), block 128, dynamic LDS
// ---------------------------------------------------------------------------
__global__ __launch_bounds__(128) void gemm_conv(const bf16_t* __restrict__ Xb,
                                                 const bf16_t* __restrict__ Wp,
                                                 const float* __restrict__ srb,
                                                 float* __restrict__ xr) {
    extern __shared__ bf16_t smem_bf[];
    bf16_t* Bb = smem_bf;
    bf16_t* Ab = smem_bf + 2 * BN * LDB;
    int tid = threadIdx.x, lane = tid & 31, wave = tid >> 5;
    int m0 = blockIdx.x * 16;
    int n_blk = blockIdx.y * BN;
    int half = (lane >> 4) & 1;

    // copy-thread A row (row = tid>>3) for each tap
    int m  = m0 + (tid >> 3);
    int b  = m / NK, pix = m % NK;
    int oh = pix / 24, ow = pix % 24;
    const bf16_t* asrc[4];
    const bf16_t* wb[4];
    #pragma unroll
    for (int t = 0; t < 4; t++) {
        int kh = t >> 1, kw = t & 1;
        int iw = 2 * ow + kw;
        int nx;
        if (oh < 24) nx = 2 * (48 * (2 * oh + kh) + iw);
        else         nx = 2 * (48 * (2 * (oh - 24) + kh) + iw) + 1;
        asrc[t] = Xb + ((size_t)b * NQ + nx) * CH;
        wb[t]   = Wp + (size_t)t * CH * CH;
    }

    v8f acc[4] = {{0,0,0,0,0,0,0,0},{0,0,0,0,0,0,0,0},{0,0,0,0,0,0,0,0},{0,0,0,0,0,0,0,0}};
    gemm_staged<4>(asrc, wb, n_blk, tid, Ab, Bb, acc);

    int n0 = n_blk + wave * 64;
    int nc = lane & 15;
    int rowbase = m0 + half * 8;
    #pragma unroll
    for (int t = 0; t < 4; t++) {
        int c = n0 + t * 16 + nc;
        float bv = srb[c];
        V8F u; u.v = acc[t];
        #pragma unroll
        for (int j = 0; j < 8; j++)
            xr[(size_t)(rowbase + j) * CH + c] = u.f[j] + bv;
    }
}

// ---------------------------------------------------------------------------
// LayerNorm over C=512, one block (256 threads) per row; fp32 in -> bf16 out
// ---------------------------------------------------------------------------
__global__ __launch_bounds__(256) void ln_kernel(const float* __restrict__ xr,
                                                 const float* __restrict__ g,
                                                 const float* __restrict__ bta,
                                                 bf16_t* __restrict__ out) {
    __shared__ float red[256];
    int row = blockIdx.x, tid = threadIdx.x;
    const float* p = xr + (size_t)row * CH;
    float a = p[tid], b = p[tid + 256];
    red[tid] = a + b;
    __syncthreads();
    for (int s = 128; s > 0; s >>= 1) {
        if (tid < s) red[tid] += red[tid + s];
        __syncthreads();
    }
    float mu = red[0] * (1.0f / 512.0f);
    __syncthreads();
    float d0 = a - mu, d1 = b - mu;
    red[tid] = d0 * d0 + d1 * d1;
    __syncthreads();
    for (int s = 128; s > 0; s >>= 1) {
        if (tid < s) red[tid] += red[tid + s];
        __syncthreads();
    }
    float rs = rsqrtf(red[0] * (1.0f / 512.0f) + 1e-5f);
    out[(size_t)row * CH + tid]       = (bf16_t)(d0 * rs * g[tid] + bta[tid]);
    out[(size_t)row * CH + tid + 256] = (bf16_t)(d1 * rs * g[tid + 256] + bta[tid + 256]);
}

// ---------------------------------------------------------------------------
// GEMM kv = xr_n @ kv_w.T + kv_b ; scatter k -> [b][h][nk][d], v -> vT[b][h][d][nk]
// grid (MK/16, 4), block 128, dynamic LDS
// ---------------------------------------------------------------------------
__global__ __launch_bounds__(128) void gemm_kv(const bf16_t* __restrict__ A,
                                               const bf16_t* __restrict__ W,
                                               const float* __restrict__ bias,
                                               bf16_t* __restrict__ kout,
                                               bf16_t* __restrict__ vtout) {
    extern __shared__ bf16_t smem_bf[];
    bf16_t* Bb = smem_bf;
    bf16_t* Ab = smem_bf + 2 * BN * LDB;
    int tid = threadIdx.x, lane = tid & 31, wave = tid >> 5;
    int m0 = blockIdx.x * 16;
    int n_blk = blockIdx.y * BN;
    int half = (lane >> 4) & 1;

    const bf16_t* asrc[1] = { A + (size_t)(m0 + (tid >> 3)) * CH };
    const bf16_t* wb[1]   = { W };
    v8f acc[4] = {{0,0,0,0,0,0,0,0},{0,0,0,0,0,0,0,0},{0,0,0,0,0,0,0,0},{0,0,0,0,0,0,0,0}};
    gemm_staged<1>(asrc, wb, n_blk, tid, Ab, Bb, acc);

    int n0 = n_blk + wave * 64;
    int b = m0 / NK;
    int nkbase = m0 % NK + half * 8;
    int nc = lane & 15;
    #pragma unroll
    for (int t = 0; t < 4; t++) {
        int c = n0 + t * 16 + nc;
        float bv = bias[c];
        V8F u; u.v = acc[t];
        if (c < CH) {
            int h = c >> 6, d = c & 63;
            #pragma unroll
            for (int j = 0; j < 8; j++)
                kout[(((size_t)b * NH + h) * NK + (nkbase + j)) * DH + d] = (bf16_t)(u.f[j] + bv);
        } else {
            int cv = c - CH;
            int h = cv >> 6, d = cv & 63;
            #pragma unroll
            for (int j = 0; j < 8; j++)
                vtout[(((size_t)b * NH + h) * DH + d) * NK + (nkbase + j)] = (bf16_t)(u.f[j] + bv);
        }
    }
}

// ---------------------------------------------------------------------------
// Attention: per block = (b,h, 16-row q tile). S(16x1152) in LDS, softmax, PV.
// block 128 (4 waves), dynamic LDS.
// ---------------------------------------------------------------------------
__global__ __launch_bounds__(128) void attn_kernel(const bf16_t* __restrict__ qbuf,
                                                   const bf16_t* __restrict__ kbuf,
                                                   const bf16_t* __restrict__ vT,
                                                   bf16_t* __restrict__ obuf) {
    extern __shared__ float smem[];
    float* Sp     = smem;                 // 16 * LDP
    float* red    = Sp + 16 * LDP;        // 128
    float* rowmax = red + 128;            // 16
    float* rowsum = rowmax + 16;          // 16

    int lane = threadIdx.x & 31, wave = threadIdx.x >> 5;
    int bh = blockIdx.y;
    int m0 = blockIdx.x * 16;
    const bf16_t* q = qbuf + (size_t)bh * NQ * DH;
    const bf16_t* k = kbuf + (size_t)bh * NK * DH;
    const bf16_t* v = vT   + (size_t)bh * DH * NK;
    int half = (lane >> 4) & 1;
    int nc = lane & 15;

    v16bf aq0 = load_a_tile(q, DH, m0, lane, 0);
    v16bf aq1 = load_a_tile(q, DH, m0, lane, 32);
    const float scale = 0.125f; // 64^-0.5

    // ---- S = q k^T * scale ----
    {
        v16bf bk0 = load_b_tile(k, DH, wave * 16, 0,  lane);
        v16bf bk1 = load_b_tile(k, DH, wave * 16, 32, lane);
        for (int nt = wave; nt < NK / 16; nt += 4) {
            int ncur  = nt * 16;
            int nnext = (nt + 4 < NK / 16) ? (nt + 4) * 16 : ncur;
            v16bf bn0 = load_b_tile(k, DH, nnext, 0,  lane);
            v16bf bn1 = load_b_tile(k, DH, nnext, 32, lane);
            v8f acc = {0,0,0,0,0,0,0,0};
            acc = wmma_bf16(aq0, bk0, acc);
            acc = wmma_bf16(aq1, bk1, acc);
            V8F u; u.v = acc;
            #pragma unroll
            for (int j = 0; j < 8; j++)
                Sp[(half * 8 + j) * LDP + ncur + nc] = u.f[j] * scale;
            bk0 = bn0; bk1 = bn1;
        }
    }
    __syncthreads();

    // ---- stable softmax: 8 threads per row ----
    int row = threadIdx.x >> 3, sub = threadIdx.x & 7;
    float mx = -1e30f;
    for (int c = sub; c < NK; c += 8) mx = fmaxf(mx, Sp[row * LDP + c]);
    red[row * 8 + sub] = mx;
    __syncthreads();
    if (sub == 0) {
        float m = red[row * 8];
        #pragma unroll
        for (int i = 1; i < 8; i++) m = fmaxf(m, red[row * 8 + i]);
        rowmax[row] = m;
    }
    __syncthreads();
    float rm = rowmax[row];
    float s = 0.0f;
    for (int c = sub; c < NK; c += 8) {
        float e = __expf(Sp[row * LDP + c] - rm);
        Sp[row * LDP + c] = e;
        s += e;
    }
    red[row * 8 + sub] = s;
    __syncthreads();
    if (sub == 0) {
        float t = 0.0f;
        #pragma unroll
        for (int i = 0; i < 8; i++) t += red[row * 8 + i];
        rowsum[row] = t;
    }
    __syncthreads();

    // ---- O = P @ v ----
    int d0 = wave * 16;
    v8f acc = {0,0,0,0,0,0,0,0};
    int ml = lane & 15;
    v16bf bv = load_b_tile(v, NK, d0, 0, lane);
    for (int kk = 0; kk < NK; kk += 32) {
        int knext = (kk + 32 < NK) ? kk + 32 : kk;
        v16bf bn = load_b_tile(v, NK, d0, knext, lane);
        const float* base = Sp + ml * LDP + kk;
        const v4f* p0 = (const v4f*)(base + half * 16);
        const v4f* p1 = (const v4f*)(base + half * 16 + 8);
        v4f f0 = p0[0], f1 = p0[1];
        v4f f2 = p1[0], f3 = p1[1];
        V16B a;
        #pragma unroll
        for (int j = 0; j < 4; j++) {
            a.e[j]      = (__bf16)f0[j];
            a.e[4 + j]  = (__bf16)f1[j];
            a.e[8 + j]  = (__bf16)f2[j];
            a.e[12 + j] = (__bf16)f3[j];
        }
        acc = wmma_bf16(a.v, bv, acc);
        bv = bn;
    }
    int b = bh >> 3, h = bh & 7;
    V8F u; u.v = acc;
    #pragma unroll
    for (int j = 0; j < 8; j++) {
        int lrow = half * 8 + j;
        float val = u.f[j] / rowsum[lrow];
        obuf[((size_t)b * NQ + (m0 + lrow)) * CH + h * DH + d0 + nc] = (bf16_t)val;
    }
}

// ---------------------------------------------------------------------------
// GEMM proj: out = O @ proj_w.T + proj_b (fp32 output to d_out)
// grid (MQ/16, 2), block 128, dynamic LDS
// ---------------------------------------------------------------------------
__global__ __launch_bounds__(128) void gemm_proj(const bf16_t* __restrict__ A,
                                                 const bf16_t* __restrict__ W,
                                                 const float* __restrict__ bias,
                                                 float* __restrict__ out) {
    extern __shared__ bf16_t smem_bf[];
    bf16_t* Bb = smem_bf;
    bf16_t* Ab = smem_bf + 2 * BN * LDB;
    int tid = threadIdx.x, lane = tid & 31, wave = tid >> 5;
    int m0 = blockIdx.x * 16;
    int n_blk = blockIdx.y * BN;
    int half = (lane >> 4) & 1;

    const bf16_t* asrc[1] = { A + (size_t)(m0 + (tid >> 3)) * CH };
    const bf16_t* wb[1]   = { W };
    v8f acc[4] = {{0,0,0,0,0,0,0,0},{0,0,0,0,0,0,0,0},{0,0,0,0,0,0,0,0},{0,0,0,0,0,0,0,0}};
    gemm_staged<1>(asrc, wb, n_blk, tid, Ab, Bb, acc);

    int n0 = n_blk + wave * 64;
    int nc = lane & 15;
    int rowbase = m0 + half * 8;
    #pragma unroll
    for (int t = 0; t < 4; t++) {
        int c = n0 + t * 16 + nc;
        float bv = bias[c];
        V8F u; u.v = acc[t];
        #pragma unroll
        for (int j = 0; j < 8; j++)
            out[(size_t)(rowbase + j) * CH + c] = u.f[j] + bv;
    }
}

// ---------------------------------------------------------------------------
// Launch
// ---------------------------------------------------------------------------
extern "C" void kernel_launch(void* const* d_in, const int* in_sizes, int n_in,
                              void* d_out, int out_size, void* d_ws, size_t ws_size,
                              hipStream_t stream) {
    const float* x      = (const float*)d_in[0];
    const float* q_w    = (const float*)d_in[1];
    const float* q_b    = (const float*)d_in[2];
    const float* kv_w   = (const float*)d_in[3];
    const float* kv_b   = (const float*)d_in[4];
    const float* sr_w   = (const float*)d_in[5];
    const float* sr_b   = (const float*)d_in[6];
    const float* ln_g   = (const float*)d_in[7];
    const float* ln_b   = (const float*)d_in[8];
    const float* proj_w = (const float*)d_in[9];
    const float* proj_b = (const float*)d_in[10];
    float* out = (float*)d_out;

    char* ws = (char*)d_ws;
    size_t off = 0;
    auto take = [&](size_t bytes) { char* p = ws + off; off += (bytes + 255) & ~(size_t)255; return p; };
    bf16_t* x_bf   = (bf16_t*)take((size_t)MQ * CH * 2);
    bf16_t* q_bf   = (bf16_t*)take((size_t)MQ * CH * 2);
    bf16_t* o_bf   = (bf16_t*)take((size_t)MQ * CH * 2);
    float*  xr_f   = (float*) take((size_t)MK * CH * 4);
    bf16_t* xr_bf  = (bf16_t*)take((size_t)MK * CH * 2);
    bf16_t* k_bf   = (bf16_t*)take((size_t)BATCH * NH * NK * DH * 2);
    bf16_t* vT_bf  = (bf16_t*)take((size_t)BATCH * NH * DH * NK * 2);
    bf16_t* qw_bf  = (bf16_t*)take((size_t)CH * CH * 2);
    bf16_t* kvw_bf = (bf16_t*)take((size_t)2 * CH * CH * 2);
    bf16_t* pw_bf  = (bf16_t*)take((size_t)CH * CH * 2);
    bf16_t* srw_bf = (bf16_t*)take((size_t)4 * CH * CH * 2);

    // 1) conversions
    {
        int n = MQ * CH;
        cvt_f32_bf16<<<(n + 255) / 256, 256, 0, stream>>>(x, x_bf, n);
        n = CH * CH;
        cvt_f32_bf16<<<(n + 255) / 256, 256, 0, stream>>>(q_w, qw_bf, n);
        cvt_f32_bf16<<<(n + 255) / 256, 256, 0, stream>>>(proj_w, pw_bf, n);
        n = 2 * CH * CH;
        cvt_f32_bf16<<<(n + 255) / 256, 256, 0, stream>>>(kv_w, kvw_bf, n);
        n = 4 * CH * CH;
        pack_srw<<<(n + 255) / 256, 256, 0, stream>>>(sr_w, srw_bf, n);
    }
    // 2) q projection
    gemm_q<<<dim3(MQ / 16, 2), 128, GEMM_LDS_BYTES, stream>>>(x_bf, qw_bf, q_b, q_bf);
    // 3) spatial-reduction conv (implicit GEMM)
    gemm_conv<<<dim3(MK / 16, 2), 128, GEMM_LDS_BYTES, stream>>>(x_bf, srw_bf, sr_b, xr_f);
    // 4) layernorm
    ln_kernel<<<MK, 256, 0, stream>>>(xr_f, ln_g, ln_b, xr_bf);
    // 5) kv projection
    gemm_kv<<<dim3(MK / 16, 4), 128, GEMM_LDS_BYTES, stream>>>(xr_bf, kvw_bf, kv_b, k_bf, vT_bf);
    // 6) attention
    size_t shmem = (size_t)(16 * LDP + 128 + 32) * sizeof(float);
    attn_kernel<<<dim3(NQ / 16, BATCH * NH), 128, shmem, stream>>>(q_bf, k_bf, vT_bf, o_bf);
    // 7) output projection
    gemm_proj<<<dim3(MQ / 16, 2), 128, GEMM_LDS_BYTES, stream>>>(o_bf, pw_bf, proj_b, out);
}
